// Concentration_61229053772314
// MI455X (gfx1250) — matrix operand order repaired and checked by reading, not compiled, not run on hardware
//
#include <hip/hip_runtime.h>
#include <hip/hip_bf16.h>
#include <math.h>

// B=64, A=32 -> BA=2048 heads; N=256 context; H=128; K=16 top-k
#define BA   2048
#define NCTX 256
#define HD   128
#define TOPK 16
#define KC   2176           // (K+1)*H
#define LSTR 132            // LDS row stride (128 + 4 pad, rotates banks)

typedef __attribute__((ext_vector_type(8)))  _Float16 v8h;
typedef __attribute__((ext_vector_type(16))) _Float16 v16h;
typedef __attribute__((ext_vector_type(8)))  float    v8f;

// ---------------------------------------------------------------------------
// Fragment helpers.  CDNA5 ISA 7.12.2 layouts (wave32):
//   A (16x32 f16): lane&15 -> M row; lane half 0 holds K {0..7,16..23},
//                  half 1 holds K {8..15,24..31}  -> two K-contiguous octets
//   B (32x16 f16): lane&15 -> N col; element e -> K = e + 8*(lane>>4) (e<8), e + 8 + 8*(lane>>4) (e>=8)
//                  -> same two K-contiguous octets as A
//   C/D (16x16 f32): VGPR r, lane -> M = r + 8*(lane>>4), N = lane&15
// With f16 operands stored K-contiguously, each fragment is 2x b128 loads.
// ---------------------------------------------------------------------------
__device__ __forceinline__ v8h ldv8h(const _Float16* p) { return *(const v8h*)p; }

__device__ __forceinline__ v16h combine16(v8h lo, v8h hi) {
  v16h r;
#pragma unroll
  for (int i = 0; i < 8; ++i) { r[i] = lo[i]; r[i + 8] = hi[i]; }
  return r;
}

// A fragment, f16 row-major source, Ar already points at (row, kk)
__device__ __forceinline__ v16h afrag_h(const _Float16* Ar, int hlf) {
  return combine16(ldv8h(Ar + hlf * 8), ldv8h(Ar + 16 + hlf * 8));
}
// B fragment, f16 n-major (Bt[n][k]) source, Br already points at (col, kk)
__device__ __forceinline__ v16h bfrag_h(const _Float16* Br, int hlf) {
  return combine16(ldv8h(Br + hlf * 8), ldv8h(Br + 16 + hlf * 8));
}
// A fragment gathered from fp32 (convert on load)
__device__ __forceinline__ v16h afrag_f32(const float* Ar, int hlf) {
  v16h r;
  const float* p0 = Ar + hlf * 8;
  const float* p1 = Ar + 16 + hlf * 8;
#pragma unroll
  for (int i = 0; i < 8; ++i) { r[i] = (_Float16)p0[i]; r[i + 8] = (_Float16)p1[i]; }
  return r;
}

#define WMMA(a, b, acc) \
  __builtin_amdgcn_wmma_f32_16x16x32_f16(false, (a), false, (b), (short)0, (acc), false, false)

__device__ __forceinline__ v8f wmma_gemm_h(const _Float16* A, int lda,
                                           const _Float16* Bt, int ldb,
                                           int mtile, int ntile, int Kdim) {
  const int lane = threadIdx.x & 31, hlf = lane >> 4, l15 = lane & 15;
  const _Float16* Arow = A + (size_t)(mtile * 16 + l15) * lda;
  const _Float16* Brow = Bt + (size_t)(ntile * 16 + l15) * ldb;
  v8f acc = {};
  for (int kk = 0; kk < Kdim; kk += 32)
    acc = WMMA(afrag_h(Arow + kk, hlf), bfrag_h(Brow + kk, hlf), acc);
  return acc;
}

__device__ __forceinline__ void store_tile_f(float* out, int ldc, int mtile,
                                             int ntile, v8f acc) {
  const int lane = threadIdx.x & 31, hlf = lane >> 4, l15 = lane & 15;
#pragma unroll
  for (int r = 0; r < 8; ++r)
    out[(size_t)(mtile * 16 + r + 8 * hlf) * ldc + ntile * 16 + l15] = acc[r];
}

__device__ __forceinline__ void store_tile_h(_Float16* out, int ldc, int mtile,
                                             int ntile, v8f acc) {
  const int lane = threadIdx.x & 31, hlf = lane >> 4, l15 = lane & 15;
#pragma unroll
  for (int r = 0; r < 8; ++r)
    out[(size_t)(mtile * 16 + r + 8 * hlf) * ldc + ntile * 16 + l15] =
        (_Float16)acc[r];
}

__device__ __forceinline__ void store_tile_bias_relu(float* out, int ldc,
                                                     int mtile, int ntile,
                                                     v8f acc, const float* bias) {
  const int lane = threadIdx.x & 31, hlf = lane >> 4, l15 = lane & 15;
  const float bb = bias[ntile * 16 + l15];
#pragma unroll
  for (int r = 0; r < 8; ++r) {
    float v = acc[r] + bb;
    out[(size_t)(mtile * 16 + r + 8 * hlf) * ldc + ntile * 16 + l15] =
        v > 0.f ? v : 0.f;
  }
}

// ------------------------------ prep kernels -------------------------------
__global__ __launch_bounds__(256) void k_cvt(const float* __restrict__ in,
                                             _Float16* __restrict__ out, int n) {
  const int i = blockIdx.x * 256 + threadIdx.x;
  if (i < n) out[i] = (_Float16)in[i];
}
// out[n][k] = in[k][n] for 128x128 (Wq, Wv used as B(k,n)=W[k][n])
__global__ __launch_bounds__(256) void k_cvtT128(const float* __restrict__ in,
                                                 _Float16* __restrict__ out) {
  const int i = blockIdx.x * 256 + threadIdx.x;   // 16384 elems
  const int n = i >> 7, k = i & 127;
  out[i] = (_Float16)in[(size_t)k * HD + n];
}

// ------------------------------ GEMM kernels -------------------------------
// f16 in, f16 out (Q = vs@Wq ; Va = s_ve@Wv)
__global__ __launch_bounds__(256) void k_gemm_h(const _Float16* __restrict__ A,
                                                const _Float16* __restrict__ Bt,
                                                _Float16* __restrict__ out) {
  v8f acc = wmma_gemm_h(A, HD, Bt, HD, blockIdx.x, threadIdx.x >> 5, HD);
  store_tile_h(out, HD, blockIdx.x, threadIdx.x >> 5, acc);
}
// f16 in, f32 out (U = Q@Wk^T, consumed by VALU streaming kernel)
__global__ __launch_bounds__(256) void k_gemm_f(const _Float16* __restrict__ A,
                                                const _Float16* __restrict__ Bt,
                                                float* __restrict__ out) {
  v8f acc = wmma_gemm_h(A, HD, Bt, HD, blockIdx.x, threadIdx.x >> 5, HD);
  store_tile_f(out, HD, blockIdx.x, threadIdx.x >> 5, acc);
}

// ------------------- bandwidth-bound streaming kernel ----------------------
// One workgroup per head: ve slice (256x128 f32, 132 KB w/ pad) is staged into
// LDS with CDNA5 async-to-LDS b128 copies (no VGPR round-trip, ASYNCcnt),
// then: compat -> masked softmax -> s_ve = score@ve -> stable top-16.
__global__ __launch_bounds__(256) void k_stream(const float* __restrict__ ve,
                                                const int* __restrict__ ve_dead,
                                                const float* __restrict__ U,
                                                _Float16* __restrict__ s_ve_h,
                                                int* __restrict__ top_idx) {
  __shared__ float lds_ve[NCTX * LSTR];
  __shared__ float lds_u[HD];
  __shared__ float sc[NCTX];
  __shared__ float red[NCTX];
  __shared__ int   ridx[NCTX];
  __shared__ float part[NCTX];

  const int ba = blockIdx.x;
  const int t  = threadIdx.x;

  // async staging: 8192 x b128, global -> LDS directly
  const float4* src = (const float4*)(ve + (size_t)ba * NCTX * HD);
  for (int i = t; i < NCTX * (HD / 4); i += 256) {
    const int n = i >> 5;            // 32 float4 per row
    const int c = i & 31;
    const unsigned lds_off =
        (unsigned)(uintptr_t)(&lds_ve[n * LSTR + c * 4]);  // low 32 = DS offset
    const unsigned long long gaddr = (unsigned long long)(src + i);
    asm volatile("global_load_async_to_lds_b128 %0, %1, off"
                 :: "v"(lds_off), "v"(gaddr) : "memory");
  }
  if (t < HD) lds_u[t] = U[(size_t)ba * HD + t];
  asm volatile("s_wait_asynccnt 0" ::: "memory");
  __syncthreads();

  // compat[n] = (ve[n] . u) / sqrt(H), masked
  float acc = 0.f;
  const float* row = &lds_ve[t * LSTR];
#pragma unroll 8
  for (int h = 0; h < HD; ++h) acc += row[h] * lds_u[h];
  const bool  dead   = ve_dead[(size_t)ba * NCTX + t] != 0;
  const float compat = acc * 0.08838834764831845f;  // 1/sqrt(128)

  // softmax: block max
  red[t] = dead ? -INFINITY : compat;
  __syncthreads();
  for (int s = NCTX / 2; s > 0; s >>= 1) {
    if (t < s) red[t] = fmaxf(red[t], red[t + s]);
    __syncthreads();
  }
  const float mx = red[0];
  __syncthreads();
  const float e = dead ? 0.f : __expf(compat - mx);
  red[t] = e;
  __syncthreads();
  for (int s = NCTX / 2; s > 0; s >>= 1) {
    if (t < s) red[t] += red[t + s];
    __syncthreads();
  }
  const float sum = red[0];
  __syncthreads();
  const float score = (sum > 0.f) ? e / sum : 0.f;  // nan_to_num for all-dead
  sc[t] = score;
  __syncthreads();

  // s_ve[h] = sum_n score[n] * ve[n][h]  (two partial halves over n)
  {
    const int h  = t & (HD - 1);
    const int n0 = (t >> 7) * (NCTX / 2);
    float a2 = 0.f;
    for (int n = n0; n < n0 + NCTX / 2; ++n) a2 += sc[n] * lds_ve[n * LSTR + h];
    part[t] = a2;
    __syncthreads();
    if (t < HD) s_ve_h[(size_t)ba * HD + t] = (_Float16)(part[t] + part[t + HD]);
  }
  __syncthreads();

  // stable top-K (smallest index wins ties, like jax.lax.top_k)
  for (int k = 0; k < TOPK; ++k) {
    red[t]  = sc[t];
    ridx[t] = t;
    __syncthreads();
    for (int s = NCTX / 2; s > 0; s >>= 1) {
      if (t < s) {
        const float a1 = red[t], b1 = red[t + s];
        if (b1 > a1 || (b1 == a1 && ridx[t + s] < ridx[t])) {
          red[t]  = b1;
          ridx[t] = ridx[t + s];
        }
      }
      __syncthreads();
    }
    const int win = ridx[0];
    if (t == 0) top_idx[(size_t)ba * TOPK + k] = win;
    __syncthreads();
    if (t == win) sc[t] = -INFINITY;
    __syncthreads();
  }
}

// v_M_final = relu([vs,Va] @ W_mot^T + b_mot)   Kdim=256, both A halves f16
__global__ __launch_bounds__(256) void k_vm(const _Float16* __restrict__ vsh,
                                            const _Float16* __restrict__ Vah,
                                            const _Float16* __restrict__ Wmoth,
                                            const float* __restrict__ bmot,
                                            float* __restrict__ outM) {
  const int mtile = blockIdx.x, ntile = threadIdx.x >> 5;
  const int lane = threadIdx.x & 31, hlf = lane >> 4, l15 = lane & 15;
  const _Float16* Ar0  = vsh + (size_t)(mtile * 16 + l15) * HD;
  const _Float16* Ar1  = Vah + (size_t)(mtile * 16 + l15) * HD;
  const _Float16* Brow = Wmoth + (size_t)(ntile * 16 + l15) * (2 * HD);
  v8f acc = {};
  for (int kk = 0; kk < 2 * HD; kk += 32) {
    const _Float16* Ar = (kk < HD) ? (Ar0 + kk) : (Ar1 + (kk - HD));
    acc = WMMA(afrag_h(Ar, hlf), bfrag_h(Brow + kk, hlf), acc);
  }
  store_tile_bias_relu(outM, HD, mtile, ntile, acc, bmot);
}

// v_C_final = relu([vs, ve[top_idx]] @ W_fwd^T + b_fwd)   Kdim=2176, gathered A
__global__ __launch_bounds__(256) void k_vc(const _Float16* __restrict__ vsh,
                                            const float* __restrict__ ve,
                                            const int* __restrict__ top_idx,
                                            const _Float16* __restrict__ Wfwdh,
                                            const float* __restrict__ bfwd,
                                            float* __restrict__ outC) {
  const int mtile = blockIdx.x, ntile = threadIdx.x >> 5;
  const int lane = threadIdx.x & 31, hlf = lane >> 4, l15 = lane & 15;
  const int mrow = mtile * 16 + l15;
  const _Float16* Ar0  = vsh + (size_t)mrow * HD;
  const _Float16* Brow = Wfwdh + (size_t)(ntile * 16 + l15) * KC;
  const int*   ti  = top_idx + (size_t)mrow * TOPK;
  const float* veb = ve + (size_t)mrow * NCTX * HD;
  v8f acc = {};
  for (int kk = 0; kk < KC; kk += 32) {
    v16h a;
    if (kk < HD) {                      // uniform branch (kk is wave-uniform)
      a = afrag_h(Ar0 + kk, hlf);
    } else {
      const int r = (kk - HD) >> 7;     // which gathered row (const per step)
      const int c = (kk - HD) & (HD - 1);
      a = afrag_f32(veb + (size_t)ti[r] * HD + c, hlf);
    }
    acc = WMMA(a, bfrag_h(Brow + kk, hlf), acc);
  }
  store_tile_bias_relu(outC, HD, mtile, ntile, acc, bfwd);
}

// ---------------------------------------------------------------------------
extern "C" void kernel_launch(void* const* d_in, const int* in_sizes, int n_in,
                              void* d_out, int out_size, void* d_ws,
                              size_t ws_size, hipStream_t stream) {
  (void)in_sizes; (void)n_in; (void)out_size; (void)ws_size;
  const float* vs   = (const float*)d_in[0];
  const float* ve   = (const float*)d_in[1];
  const int*   ved  = (const int*)d_in[2];
  const float* Wq   = (const float*)d_in[3];
  const float* Wk   = (const float*)d_in[4];
  const float* Wv   = (const float*)d_in[5];
  const float* Wmot = (const float*)d_in[6];
  const float* bmot = (const float*)d_in[7];
  const float* Wfwd = (const float*)d_in[8];
  const float* bfwd = (const float*)d_in[9];

  float* outC = (float*)d_out;                     // v_C_final [2048,128]
  float* outM = (float*)d_out + (size_t)BA * HD;   // v_M_final [2048,128]

  // workspace carve-up (all chunks 256B-aligned)
  char* w = (char*)d_ws;
  _Float16* vsh   = (_Float16*)w; w += (size_t)BA * HD * 2;
  _Float16* Qh    = (_Float16*)w; w += (size_t)BA * HD * 2;
  _Float16* sveh  = (_Float16*)w; w += (size_t)BA * HD * 2;
  _Float16* Vah   = (_Float16*)w; w += (size_t)BA * HD * 2;
  float*    U     = (float*)w;    w += (size_t)BA * HD * 4;
  int*      tidx  = (int*)w;      w += (size_t)BA * TOPK * 4;
  _Float16* WqTh  = (_Float16*)w; w += (size_t)HD * HD * 2;
  _Float16* Wkh   = (_Float16*)w; w += (size_t)HD * HD * 2;
  _Float16* WvTh  = (_Float16*)w; w += (size_t)HD * HD * 2;
  _Float16* Wmoth = (_Float16*)w; w += (size_t)HD * 2 * HD * 2;
  _Float16* Wfwdh = (_Float16*)w; w += (size_t)HD * KC * 2;

  const dim3 blk(256);
  // one-shot f16 prep (all tiny / L2-resident)
  k_cvt<<<dim3((BA * HD + 255) / 256), blk, 0, stream>>>(vs, vsh, BA * HD);
  k_cvtT128<<<dim3(HD * HD / 256), blk, 0, stream>>>(Wq, WqTh);
  k_cvt<<<dim3(HD * HD / 256), blk, 0, stream>>>(Wk, Wkh, HD * HD);
  k_cvtT128<<<dim3(HD * HD / 256), blk, 0, stream>>>(Wv, WvTh);
  k_cvt<<<dim3(HD * 2 * HD / 256), blk, 0, stream>>>(Wmot, Wmoth, HD * 2 * HD);
  k_cvt<<<dim3(HD * KC / 256), blk, 0, stream>>>(Wfwd, Wfwdh, HD * KC);

  // pipeline
  k_gemm_h<<<dim3(BA / 16), blk, 0, stream>>>(vsh, WqTh, Qh);      // Q = vs@Wq
  k_gemm_f<<<dim3(BA / 16), blk, 0, stream>>>(Qh, Wkh, U);         // U = Q@Wk^T
  k_stream<<<dim3(BA), blk, 0, stream>>>(ve, ved, U, sveh, tidx);  // softmax etc
  k_gemm_h<<<dim3(BA / 16), blk, 0, stream>>>(sveh, WvTh, Vah);    // Va
  k_vm<<<dim3(BA / 16), blk, 0, stream>>>(vsh, Vah, Wmoth, bmot, outM);
  k_vc<<<dim3(BA / 16), blk, 0, stream>>>(vsh, ve, tidx, Wfwdh, bfwd, outC);
}
